// WeightOnlyInt8Linear_20830591385798
// MI455X (gfx1250) — compile-verified, matrix operand order checked by reading, not run on hardware
//
#include <hip/hip_runtime.h>
#include <hip/hip_bf16.h>
#include <stdint.h>

typedef __attribute__((ext_vector_type(16))) __bf16 v16bf;
typedef __attribute__((ext_vector_type(8)))  __bf16 v8bf;
typedef __attribute__((ext_vector_type(8)))  float  v8f;
typedef __attribute__((ext_vector_type(4)))  int    v4i;
typedef __attribute__((ext_vector_type(4)))  unsigned int u32x4;
typedef __attribute__((ext_vector_type(8)))  int    i32x8;

#define K_DIM 4096
#define N_DIM 11008
#define M_DIM 8192
#define BM 128
#define BN 128
#define BK 64
#define NT (K_DIM / BK)   /* 64 K-tiles */
#define AROW 72           /* A LDS row stride in bf16 = 144 B (TDM pad: 16B per 128B row) */
#define ABYTES (BM * AROW * 2)
#define BROWP 40          /* B LDS row stride in bf16 = 80 B */

union FragU { v16bf v; v8bf h[2]; };

__device__ static inline unsigned lds_off_u32(const void* p) {
    // AS3->flat addrspacecast puts the DS byte address in the low 32 bits
    return (unsigned)(uintptr_t)p;
}

__global__ __launch_bounds__(256)
void wint8_gemm_bf16(const __bf16* __restrict__ X,
                     const int8_t* __restrict__ Wq,
                     const __hip_bfloat16* __restrict__ S,
                     __hip_bfloat16* __restrict__ Out)
{
    __shared__ __align__(16) __bf16 Atile[2][BM * AROW];    // double-buffered TDM dest
    __shared__ __align__(16) __bf16 Btile[2][2][BN][BROWP]; // [buf][kstep][n][k in 32]

    const int tid  = threadIdx.x;
    const int lane = tid & 31;
    const int wave = tid >> 5;     // 0..7
    const int wm   = wave & 3;     // wave M position (x32 rows)
    const int wn   = wave >> 2;    // wave N position (x64 cols)
    const int lm   = lane & 15;    // row (A) / col (B) within 16x16 frag
    const int lh   = lane >> 4;    // lane half selects K sub-runs

    const int n0 = blockIdx.x * BN;
    const int m0 = blockIdx.y * BM;

    v8f acc[2][4];
#pragma unroll
    for (int i = 0; i < 2; ++i)
#pragma unroll
        for (int j = 0; j < 4; ++j) { v8f z = {}; acc[i][j] = z; }

    // B staging assignment: thread handles rows br, br+64, 16-int8 chunk bc
    const int br = tid >> 2;
    const int bc = tid & 3;

    // ---- TDM descriptor group1 (loop-invariant) ----
    // data_size=1(2B), pad_enable, pad_interval=4 (32 dwords), pad_amount=3 (4 dwords)
    i32x8 g1;
    g1[0] = (1 << 16) | (1 << 20) | (4 << 22) | (3 << 25);
    g1[1] = (int)((unsigned)(K_DIM & 0xFFFF) << 16);        // tensor_dim0[15:0] @ bits63:48
    g1[2] = (int)((K_DIM >> 16) & 0xFFFF) | (BM << 16);     // tensor_dim0[31:16] | tensor_dim1[15:0]
    g1[3] = (int)((unsigned)BK << 16);                      // tile_dim0 @ bits127:112
    g1[4] = BM;                                             // tile_dim1 @ bits143:128
    g1[5] = K_DIM;                                          // tensor_dim0_stride low32
    g1[6] = 0;
    g1[7] = 0;
    const v4i  zero4 = {0, 0, 0, 0};
    const i32x8 zero8 = {0, 0, 0, 0, 0, 0, 0, 0};
    const unsigned aLds = lds_off_u32(&Atile[0][0]);

    auto issue_tdm = [&](int kt, int buf) {
        const uint64_t ga = (uint64_t)(uintptr_t)(X + (size_t)m0 * K_DIM + kt * BK);
        u32x4 g0;
        g0.x = 1u;                                                // count=1
        g0.y = aLds + (unsigned)buf * ABYTES;                     // lds_addr
        g0.z = (unsigned)ga;                                      // global_addr[31:0]
        g0.w = (unsigned)((ga >> 32) & 0x01FFFFFFu) | (2u << 30); // addr[56:32] | type=2
        __builtin_amdgcn_tensor_load_to_lds(g0, g1, zero4, zero4, zero8, 0);
    };

    // ---- pipeline prologue: start A-tile 0 DMA, preload B-tile 0 into VGPRs ----
    if (wave == 0) issue_tdm(0, 0);
    v4i braw[2];
#pragma unroll
    for (int rr = 0; rr < 2; ++rr)
        braw[rr] = *(const v4i*)(Wq + (size_t)(n0 + br + rr * 64) * K_DIM + bc * 16);

    for (int kt = 0; kt < NT; ++kt) {
        const int cur = kt & 1;
        const int nxt = cur ^ 1;
        const bool more = (kt + 1) < NT;

        // ---- dequant + store this tile's B (raw int8 already in VGPRs) ----
#pragma unroll
        for (int rr = 0; rr < 2; ++rr) {
            const int r = br + rr * 64;
            const int8_t* rb = (const int8_t*)&braw[rr];
            __bf16 tmp[16];
#pragma unroll
            for (int i = 0; i < 16; ++i) tmp[i] = (__bf16)(float)rb[i];
            const int ks = bc >> 1;          // which 32-wide kstep
            const int cb = (bc & 1) * 16;    // col base within kstep
            *(v8bf*)&Btile[cur][ks][r][cb]     = *(const v8bf*)&tmp[0];
            *(v8bf*)&Btile[cur][ks][r][cb + 8] = *(const v8bf*)&tmp[8];
        }

        // ---- kick off next tile: A via TDM, B raw into VGPRs ----
        if (more) {
            if (wave == 0) issue_tdm(kt + 1, nxt);
            const int k1 = (kt + 1) * BK;
#pragma unroll
            for (int rr = 0; rr < 2; ++rr)
                braw[rr] = *(const v4i*)(Wq + (size_t)(n0 + br + rr * 64) * K_DIM + k1 + bc * 16);
            if (kt + 2 < NT)
                __builtin_prefetch(Wq + (size_t)(n0 + br) * K_DIM + k1 + BK, 0, 0);
        }

        // TDM completes in order per wave: <=1 outstanding means tile `kt` landed
        if (wave == 0) {
            if (more) __builtin_amdgcn_s_wait_tensorcnt(1);
            else      __builtin_amdgcn_s_wait_tensorcnt(0);
        }
        __syncthreads();

        // ---- compute: 2 ksteps x (2 A-frags x 4 B-frags) WMMAs ----
#pragma unroll
        for (int s = 0; s < 2; ++s) {
            FragU a[2], b[4];
#pragma unroll
            for (int i = 0; i < 2; ++i) {
                const __bf16* p = &Atile[cur][(wm * 32 + i * 16 + lm) * AROW + s * 32];
                a[i].h[0] = *(const v8bf*)(p + lh * 8);
                a[i].h[1] = *(const v8bf*)(p + 16 + lh * 8);
            }
#pragma unroll
            for (int j = 0; j < 4; ++j) {
                const __bf16* p = &Btile[cur][s][wn * 64 + j * 16 + lm][0];
                b[j].h[0] = *(const v8bf*)(p + lh * 8);
                b[j].h[1] = *(const v8bf*)(p + 16 + lh * 8);
            }
#pragma unroll
            for (int i = 0; i < 2; ++i)
#pragma unroll
                for (int j = 0; j < 4; ++j)
                    acc[i][j] = __builtin_amdgcn_wmma_f32_16x16x32_bf16(
                        false, a[i].v, false, b[j].v,
                        (short)0, acc[i][j], false, false);
        }
        __syncthreads();
    }

    // ---- epilogue: scale + bf16 store ----
#pragma unroll
    for (int j = 0; j < 4; ++j) {
        const int n = n0 + wn * 64 + j * 16 + lm;
        const float sc = __bfloat162float(S[n]);
#pragma unroll
        for (int i = 0; i < 2; ++i) {
            const int mb = m0 + wm * 32 + i * 16 + lh * 8;  // VGPR v holds M = v + 8*lh
#pragma unroll
            for (int v = 0; v < 8; ++v) {
                Out[(size_t)(mb + v) * N_DIM + n] = __float2bfloat16(acc[i][j][v] * sc);
            }
        }
    }
}

extern "C" void kernel_launch(void* const* d_in, const int* in_sizes, int n_in,
                              void* d_out, int out_size, void* d_ws, size_t ws_size,
                              hipStream_t stream) {
    const __bf16*         x = (const __bf16*)d_in[0];
    const int8_t*         w = (const int8_t*)d_in[1];
    const __hip_bfloat16* s = (const __hip_bfloat16*)d_in[2];
    __hip_bfloat16*     out = (__hip_bfloat16*)d_out;

    dim3 grid(N_DIM / BN, M_DIM / BM, 1);
    wint8_gemm_bf16<<<grid, dim3(256, 1, 1), 0, stream>>>(x, w, s, out);
}
